// LocalMHA_32538672234962
// MI455X (gfx1250) — compile-verified, matrix operand order checked by reading, not run on hardware
//
#include <hip/hip_runtime.h>
#include <hip/hip_bf16.h>

// ---------------------------------------------------------------------------
// Local windowed MHA forward for MI455X (gfx1250), wave32 + WMMA f16->f32.
//   b=2, n=8192, DIM=1024, HEADS=16, DHEAD=64, WINDOW=128, look_backward=1
// Pipeline: LN(pack A-frag) -> pack weights(B-frag) -> QKV GEMM (WMMA, scatter
// q/kT/v frags) -> per-window attention (WMMA QK^T, softmax, WMMA PV) ->
// output GEMM (WMMA) -> f32 out.
// ---------------------------------------------------------------------------

typedef __attribute__((ext_vector_type(16))) _Float16 v16h;
typedef __attribute__((ext_vector_type(8)))  float    v8f;
typedef _Float16 half_t;

#define HEADS  16
#define DIM    1024
#define DHEAD  64
#define WIN    128
#define NTOK   8192
#define BATCH  2
#define ROWS   (BATCH * NTOK)   // 16384
#define NQKV   (3 * DIM)        // 3072
#define NWIN   (NTOK / WIN)     // 64

// Workspace layout (in halfs). All offsets keep >=32B alignment.
static constexpr size_t OFF_XNA  = 0;                       // 16384x1024 A-frag (reused as O A-frag)
static constexpr size_t OFF_WQKV = OFF_XNA  + 16777216;     // 1024x3072 B-frag
static constexpr size_t OFF_WOUT = OFF_WQKV + 3145728;      // 1024x1024 B-frag
static constexpr size_t OFF_QA   = OFF_WOUT + 1048576;      // per (b,h): 8192x64 A-frag
static constexpr size_t OFF_KTB  = OFF_QA   + 16777216;     // per (b,h): B-frag (d x padded keypos), 520 jtiles
static constexpr size_t OFF_VB   = OFF_KTB  + 17039360;     // per (b,h): B-frag (padded keypos x d), 260 ktiles
// total = 71,827,456 halfs = ~137 MB

static constexpr size_t QA_BH  = 524288;  // 512 mtiles * 2 ktiles * 512
static constexpr size_t KTB_BH = 532480;  // 520 jtiles * 2 dtiles * 512
static constexpr size_t VB_BH  = 532480;  // 4 ntiles * 260 ktiles * 512

// --- WMMA fragment addressing (element granularity) ------------------------
// A-frag (MxK): tile(mt,kt) base=((mt*Kt+kt)<<9); within tile, element (m,k):
//   lane = (m&15) | ((k>>3 & 1)<<4), halfidx = (k&7) | ((k>>4 & 1)<<3)
__device__ __forceinline__ size_t a_off(int kt32, int m, int k) {
  int kk   = k & 31;
  int lane = (m & 15) | (((kk >> 3) & 1) << 4);
  int hidx = (kk & 7) | (((kk >> 4) & 1) << 3);
  return ((size_t)((m >> 4) * kt32 + (k >> 5)) << 9) + (lane << 4) + hidx;
}
// B-frag (KxN): tile(nt,kt) base=((nt*Kt+kt)<<9); element (k,n) mirrors A.
__device__ __forceinline__ size_t b_off(int kt32, int k, int n) {
  int kk   = k & 31;
  int lane = (n & 15) | (((kk >> 3) & 1) << 4);
  int hidx = (kk & 7) | (((kk >> 4) & 1) << 3);
  return ((size_t)((n >> 4) * kt32 + (k >> 5)) << 9) + (lane << 4) + hidx;
}

__device__ __forceinline__ v16h load_frag(const half_t* p) {
  union { uint4 u[2]; v16h v; } t;
  t.u[0] = *(const uint4*)(p);
  t.u[1] = *(const uint4*)(p + 8);
  return t.v;
}

__device__ __forceinline__ v8f zero8() {
  v8f z;
#pragma unroll
  for (int e = 0; e < 8; ++e) z[e] = 0.0f;
  return z;
}

__device__ __forceinline__ v8f wmma_f16(v16h a, v16h b, v8f c) {
  return __builtin_amdgcn_wmma_f32_16x16x32_f16(false, a, false, b, (short)0, c,
                                                false, false);
}

// ---------------------------------------------------------------------------
// Kernel 1: LayerNorm one row per block; write normalized row as f16 A-frags.
// ---------------------------------------------------------------------------
__global__ __launch_bounds__(256)
void ln_pack_a(const float* __restrict__ x, const float* __restrict__ g,
               const float* __restrict__ be, half_t* __restrict__ xnA) {
  const int row = blockIdx.x;
  const int t   = threadIdx.x;
  float4 v = ((const float4*)(x + (size_t)row * DIM))[t];
  float s  = v.x + v.y + v.z + v.w;
  float s2 = v.x * v.x + v.y * v.y + v.z * v.z + v.w * v.w;
  __shared__ float rs[256], rq[256];
  rs[t] = s; rq[t] = s2;
  __syncthreads();
  for (int o = 128; o > 0; o >>= 1) {
    if (t < o) { rs[t] += rs[t + o]; rq[t] += rq[t + o]; }
    __syncthreads();
  }
  const float mu   = rs[0] * (1.0f / DIM);
  const float var  = rq[0] * (1.0f / DIM) - mu * mu;
  const float rstd = rsqrtf(var + 1e-5f);
  float4 gv = ((const float4*)g)[t];
  float4 bv = ((const float4*)be)[t];
  float o0 = (v.x - mu) * rstd * gv.x + bv.x;
  float o1 = (v.y - mu) * rstd * gv.y + bv.y;
  float o2 = (v.z - mu) * rstd * gv.z + bv.z;
  float o3 = (v.w - mu) * rstd * gv.w + bv.w;
  const int c0 = t * 4;
  xnA[a_off(32, row, c0 + 0)] = (half_t)o0;
  xnA[a_off(32, row, c0 + 1)] = (half_t)o1;
  xnA[a_off(32, row, c0 + 2)] = (half_t)o2;
  xnA[a_off(32, row, c0 + 3)] = (half_t)o3;
}

// ---------------------------------------------------------------------------
// Kernel 2: pack a row-major KxN f32 weight matrix into f16 B-frags.
// ---------------------------------------------------------------------------
__global__ __launch_bounds__(256)
void pack_b(const float* __restrict__ w, half_t* __restrict__ dst, int K, int N) {
  const int idx = blockIdx.x * 256 + threadIdx.x;
  const int k = idx / N, n = idx - k * N;
  dst[b_off(K >> 5, k, n)] = (half_t)w[idx];
}

// ---------------------------------------------------------------------------
// Kernel 3: zero the one-chunk (128 keypos) lookback pad of kT / v frags.
// ---------------------------------------------------------------------------
__global__ __launch_bounds__(256)
void zero_pad(half_t* __restrict__ kTB, half_t* __restrict__ vB) {
  const int tid = blockIdx.x * 256 + threadIdx.x;  // 65536 units of 8 halfs
  const int bh  = tid >> 11;
  const int rem = tid & 2047;
  const uint4 z = make_uint4(0u, 0u, 0u, 0u);
  if (rem < 1024) {
    // kT pad: jtiles 0..7 -> frags 0..15 contiguous (8192 halfs)
    *(uint4*)(kTB + (size_t)bh * KTB_BH + (size_t)rem * 8) = z;
  } else {
    // v pad: frag (nt*260 + kt), nt=0..3, kt=0..3
    const int r2   = rem - 1024;
    const int frag = r2 >> 6, unit = r2 & 63;
    const int nt = frag >> 2, kt = frag & 3;
    *(uint4*)(vB + (size_t)bh * VB_BH + (((size_t)(nt * 260 + kt)) << 9) +
              (size_t)unit * 8) = z;
  }
}

// ---------------------------------------------------------------------------
// Shared GEMM core: 4 waves/block, each wave a 64x64 tile (4x4 WMMA tiles).
// ---------------------------------------------------------------------------
__device__ __forceinline__ void gemm_tile64(const half_t* __restrict__ A,
                                            const half_t* __restrict__ B,
                                            int mt0, int nt0, int Kt,
                                            v8f c[4][4]) {
  const int lane = threadIdx.x & 31;
#pragma unroll
  for (int i = 0; i < 4; ++i)
#pragma unroll
    for (int j = 0; j < 4; ++j) c[i][j] = zero8();
#pragma unroll 1
  for (int tk = 0; tk < Kt; ++tk) {
    v16h a[4], b[4];
#pragma unroll
    for (int i = 0; i < 4; ++i)
      a[i] = load_frag(A + (((size_t)(mt0 + i) * Kt + tk) << 9) + lane * 16);
#pragma unroll
    for (int j = 0; j < 4; ++j)
      b[j] = load_frag(B + (((size_t)(nt0 + j) * Kt + tk) << 9) + lane * 16);
#pragma unroll
    for (int i = 0; i < 4; ++i)
#pragma unroll
      for (int j = 0; j < 4; ++j) c[i][j] = wmma_f16(a[i], b[j], c[i][j]);
  }
}

// ---------------------------------------------------------------------------
// Kernel 4: QKV GEMM. C=(16384x3072). Scatter epilogue into q/kT/v frags.
// ---------------------------------------------------------------------------
__global__ __launch_bounds__(128)
void gemm_qkv(const half_t* __restrict__ A, const half_t* __restrict__ B,
              half_t* __restrict__ qA, half_t* __restrict__ kTB,
              half_t* __restrict__ vB) {
  const int lane = threadIdx.x & 31, wid = threadIdx.x >> 5;
  const int wm = wid >> 1, wn = wid & 1;
  const int mt0 = blockIdx.y * 8 + wm * 4;
  const int nt0 = blockIdx.x * 8 + wn * 4;
  v8f c[4][4];
  gemm_tile64(A, B, mt0, nt0, 32, c);

  const int rbase = (lane >> 4) << 3;
  const int ncol  = lane & 15;
#pragma unroll
  for (int i = 0; i < 4; ++i) {
#pragma unroll
    for (int j = 0; j < 4; ++j) {
      const int gc   = (nt0 + j) * 16 + ncol;   // 0..3071
      const int part = gc >> 10;                // 0=q 1=k 2=v
      const int cc   = gc & 1023;
      const int h    = cc >> 6, d = cc & 63;
#pragma unroll
      for (int v = 0; v < 8; ++v) {
        const int grow = (mt0 + i) * 16 + v + rbase;
        const int bb   = grow >> 13, ntok = grow & (NTOK - 1);
        const int bh   = bb * HEADS + h;
        const float val = c[i][j][v];
        if (part == 0) {
          qA[(size_t)bh * QA_BH + a_off(2, ntok, d)] = (half_t)(val * 0.125f);
        } else if (part == 1) {
          kTB[(size_t)bh * KTB_BH + b_off(2, d, ntok + WIN)] = (half_t)val;
        } else {
          vB[(size_t)bh * VB_BH + b_off(260, ntok + WIN, d)] = (half_t)val;
        }
      }
    }
  }
}

// ---------------------------------------------------------------------------
// Kernel 5: local attention. One block per (b,h,window); 8 waves; each wave
// owns a 16-row query strip: QK^T (32 WMMA), mask+softmax, PV (32 WMMA).
// ---------------------------------------------------------------------------
__global__ __launch_bounds__(256)
void attn_kernel(const half_t* __restrict__ qA, const half_t* __restrict__ kTB,
                 const half_t* __restrict__ vB, half_t* __restrict__ oA) {
  extern __shared__ half_t s_attn[];  // 8 waves * 16 rows * 256 cols f16 = 64KB
  const int lane = threadIdx.x & 31, mt = threadIdx.x >> 5;
  const int wi = blockIdx.x & (NWIN - 1);
  const int h  = (blockIdx.x >> 6) & (HEADS - 1);
  const int b  = blockIdx.x >> 10;
  const int bh = b * HEADS + h;
  const half_t* qbase = qA  + (size_t)bh * QA_BH;
  const half_t* kbase = kTB + (size_t)bh * KTB_BH;
  const half_t* vbase = vB  + (size_t)bh * VB_BH;

  const v16h aq0 = load_frag(qbase + (((size_t)(wi * 8 + mt) * 2 + 0) << 9) + lane * 16);
  const v16h aq1 = load_frag(qbase + (((size_t)(wi * 8 + mt) * 2 + 1) << 9) + lane * 16);

  // --- sim = q @ k^T : 16x256 strip as 16 WMMA C tiles -----------------
  v8f c[16];
#pragma unroll
  for (int nt = 0; nt < 16; ++nt) {
    const v16h b0 = load_frag(kbase + (((size_t)(wi * 8 + nt) * 2 + 0) << 9) + lane * 16);
    const v16h b1 = load_frag(kbase + (((size_t)(wi * 8 + nt) * 2 + 1) << 9) + lane * 16);
    v8f acc = zero8();
    acc = wmma_f16(aq0, b0, acc);
    acc = wmma_f16(aq1, b1, acc);
    c[nt] = acc;
  }

  // --- causal local mask + softmax per row, pack attn f16 into LDS -----
  const int rbase = (lane >> 4) << 3;
  const int ncol  = lane & 15;
#pragma unroll
  for (int j = 0; j < 8; ++j) {
    const int q_pos = wi * WIN + mt * 16 + j + rbase;
    float m = -3.0e38f;
#pragma unroll
    for (int nt = 0; nt < 16; ++nt) {
      const int k_pos = wi * WIN - WIN + nt * 16 + ncol;
      float v = c[nt][j];
      const bool bad = (k_pos < 0) | (q_pos < k_pos) | (q_pos > k_pos + WIN);
      v = bad ? -3.0e38f : v;
      c[nt][j] = v;
      m = fmaxf(m, v);
    }
    m = fmaxf(m, __shfl_xor(m, 1));
    m = fmaxf(m, __shfl_xor(m, 2));
    m = fmaxf(m, __shfl_xor(m, 4));
    m = fmaxf(m, __shfl_xor(m, 8));
    float s = 0.0f;
#pragma unroll
    for (int nt = 0; nt < 16; ++nt) {
      const float e = __expf(c[nt][j] - m);
      c[nt][j] = e;
      s += e;
    }
    s += __shfl_xor(s, 1);
    s += __shfl_xor(s, 2);
    s += __shfl_xor(s, 4);
    s += __shfl_xor(s, 8);
    const float inv = 1.0f / s;
    const int r = j + rbase;
    half_t* dst = s_attn + mt * 4096 + r * 256 + ncol;
#pragma unroll
    for (int nt = 0; nt < 16; ++nt) dst[nt * 16] = (half_t)(c[nt][j] * inv);
  }
  __syncthreads();

  // --- out = attn @ v : re-fragment attn from LDS as A operand ---------
  v8f oc[4];
#pragma unroll
  for (int nt = 0; nt < 4; ++nt) oc[nt] = zero8();
  const half_t* srow = s_attn + mt * 4096;
  const int arow = lane & 15;
  const int koff = (lane >> 4) ? 8 : 0;
#pragma unroll
  for (int kt = 0; kt < 8; ++kt) {
    union { uint4 u[2]; v16h v; } ua;
    const int cb = kt * 32 + koff;
    ua.u[0] = *(const uint4*)(srow + arow * 256 + cb);
    ua.u[1] = *(const uint4*)(srow + arow * 256 + cb + 16);
    const v16h af = ua.v;
#pragma unroll
    for (int nt = 0; nt < 4; ++nt) {
      const v16h bf =
          load_frag(vbase + (((size_t)(nt * 260 + wi * 4 + kt)) << 9) + lane * 16);
      oc[nt] = wmma_f16(af, bf, oc[nt]);
    }
  }

  // --- scatter O strip as A-frags of the (16384 x 1024) output matrix --
  const int m0 = b * NTOK + wi * WIN + mt * 16;
#pragma unroll
  for (int nt = 0; nt < 4; ++nt) {
    const int hd = h * DHEAD + nt * 16 + ncol;
#pragma unroll
    for (int v = 0; v < 8; ++v) {
      const int r = v + rbase;
      oA[a_off(32, m0 + r, hd)] = (half_t)oc[nt][v];
    }
  }
}

// ---------------------------------------------------------------------------
// Kernel 6: output GEMM, f32 row-major result.
// ---------------------------------------------------------------------------
__global__ __launch_bounds__(128)
void gemm_out(const half_t* __restrict__ A, const half_t* __restrict__ B,
              float* __restrict__ out) {
  const int lane = threadIdx.x & 31, wid = threadIdx.x >> 5;
  const int wm = wid >> 1, wn = wid & 1;
  const int mt0 = blockIdx.y * 8 + wm * 4;
  const int nt0 = blockIdx.x * 8 + wn * 4;
  v8f c[4][4];
  gemm_tile64(A, B, mt0, nt0, 32, c);
  const int rbase = (lane >> 4) << 3;
  const int ncol  = lane & 15;
#pragma unroll
  for (int i = 0; i < 4; ++i)
#pragma unroll
    for (int j = 0; j < 4; ++j)
#pragma unroll
      for (int v = 0; v < 8; ++v) {
        const int row = (mt0 + i) * 16 + v + rbase;
        const int col = (nt0 + j) * 16 + ncol;
        out[(size_t)row * DIM + col] = c[i][j][v];
      }
}

// ---------------------------------------------------------------------------
extern "C" void kernel_launch(void* const* d_in, const int* in_sizes, int n_in,
                              void* d_out, int out_size, void* d_ws,
                              size_t ws_size, hipStream_t stream) {
  (void)in_sizes; (void)n_in; (void)out_size; (void)ws_size;
  const float* x    = (const float*)d_in[0];
  const float* g    = (const float*)d_in[1];
  const float* be   = (const float*)d_in[2];
  const float* wqkv = (const float*)d_in[3];
  const float* wout = (const float*)d_in[4];

  half_t* ws    = (half_t*)d_ws;
  half_t* xnA   = ws + OFF_XNA;   // reused as O A-frag after attention
  half_t* wqkvB = ws + OFF_WQKV;
  half_t* woutB = ws + OFF_WOUT;
  half_t* qA    = ws + OFF_QA;
  half_t* kTB   = ws + OFF_KTB;
  half_t* vB    = ws + OFF_VB;

  ln_pack_a<<<ROWS, 256, 0, stream>>>(x, g, be, xnA);
  pack_b<<<(DIM * NQKV) / 256, 256, 0, stream>>>(wqkv, wqkvB, DIM, NQKV);
  pack_b<<<(DIM * DIM) / 256, 256, 0, stream>>>(wout, woutB, DIM, DIM);
  zero_pad<<<256, 256, 0, stream>>>(kTB, vB);
  gemm_qkv<<<dim3(NQKV / 128, ROWS / 128), 128, 0, stream>>>(xnA, wqkvB, qA, kTB, vB);
  attn_kernel<<<BATCH * HEADS * NWIN, 256, 65536, stream>>>(qA, kTB, vB, xnA);
  gemm_out<<<dim3(DIM / 128, ROWS / 128), 128, 0, stream>>>(xnA, woutB, (float*)d_out);
}
